// NMTDecoder_14070312862429
// MI455X (gfx1250) — matrix operand order repaired and checked by reading, not compiled, not run on hardware
//
#include <hip/hip_runtime.h>
#include <cstdint>
#include <cstddef>

// Problem dims (compile-time)
#define Bsz 32
#define Ssz 64
#define Tsz 64
#define Vsz 32000
#define Esz 256
#define Hsz 512
#define G3H 1536      // 3*H
#define EK  256       // emb GEMM K
#define PK  1024      // 2*H  (classifier K)
#define PM  2048      // T*B  (GEMM M)

typedef __attribute__((ext_vector_type(16))) __bf16 v16bf;
typedef __attribute__((ext_vector_type(8)))  __bf16 v8bf;
typedef __attribute__((ext_vector_type(8)))  float  v8f;

static __device__ __forceinline__ unsigned short f2bf(float f) {
  unsigned u = __float_as_uint(f);
  unsigned r = u + 0x7FFFu + ((u >> 16) & 1u);   // round-to-nearest-even
  return (unsigned short)(r >> 16);
}

static __device__ __forceinline__ v8f wmma_bf16(v16bf a, v16bf b, v8f c) {
  return __builtin_amdgcn_wmma_f32_16x16x32_bf16(false, a, false, b, (short)0,
                                                 c, false, false);
}

// A-fragment: lane's 16 halves = [p, p+8) and [p+8, p+16) (p already at k+16g)
static __device__ __forceinline__ v16bf load_afrag(const unsigned short* p) {
  v8bf lo = *reinterpret_cast<const v8bf*>(p);
  v8bf hi = *reinterpret_cast<const v8bf*>(p + 8);
  return __builtin_shufflevector(lo, hi, 0, 1, 2, 3, 4, 5, 6, 7,
                                 8, 9, 10, 11, 12, 13, 14, 15);
}

// ---------------- prep kernels ----------------

__global__ void k_f32_to_bf16(const float* __restrict__ src,
                              unsigned short* __restrict__ dst, size_t n) {
  size_t i = (size_t)blockIdx.x * blockDim.x + threadIdx.x;
  size_t stride = (size_t)gridDim.x * blockDim.x;
  for (; i < n; i += stride) dst[i] = f2bf(src[i]);
}

// dst[(c-c0)*rows + r] = src[r*cols + c]  for c in [c0, c0+nc)  (fp32 sub-transpose)
__global__ void k_transpose_sub(const float* __restrict__ src,
                                float* __restrict__ dst,
                                int rows, int cols, int c0, int nc) {
  int n = rows * nc;
  for (int i = blockIdx.x * blockDim.x + threadIdx.x; i < n;
       i += gridDim.x * blockDim.x) {
    int r = i / nc, cc = i - r * nc;
    dst[cc * rows + r] = src[r * cols + c0 + cc];
  }
}

// dst[r*nc + cc] = bf16(src[r*cols + c0+cc])   (bf16 sub-matrix copy, row-major)
__global__ void k_conv_sub(const float* __restrict__ src,
                           unsigned short* __restrict__ dst,
                           int rows, int cols, int c0, int nc) {
  int n = rows * nc;
  for (int i = blockIdx.x * blockDim.x + threadIdx.x; i < n;
       i += gridDim.x * blockDim.x) {
    int r = i / nc, cc = i - r * nc;
    dst[i] = f2bf(src[r * cols + c0 + cc]);
  }
}

// Aemb[m,k] = bf16(emb[y(m), k]),  m = t*B + b,  y = tgt[b*T + t]
__global__ void k_gather_emb(const float* __restrict__ emb,
                             const int* __restrict__ tgt,
                             unsigned short* __restrict__ Aemb) {
  const int n = PM * EK;
  for (int i = blockIdx.x * blockDim.x + threadIdx.x; i < n;
       i += gridDim.x * blockDim.x) {
    int m = i >> 8, k = i & 255;
    int bb = m & 31, tt = m >> 5;
    int y = tgt[bb * Tsz + tt];
    Aemb[i] = f2bf(emb[(size_t)y * EK + k]);
  }
}

// ---------------- GI_E = Aemb @ w_ih[:, :E].T + b_ih  (bf16 WMMA) ----------------
// block 256 thr (8 waves: 2M x 4N); wave tile 32M x 64N; block 64M x 256N
// grid = (1536/256, 2048/64) = (6, 32); K = 256

__global__ __launch_bounds__(256) void k_gie_gemm(
    const unsigned short* __restrict__ Aemb,   // [2048,256] bf16
    const unsigned short* __restrict__ w1bf,   // [1536,256] bf16
    const float* __restrict__ b_ih,            // [1536]
    float* __restrict__ gie)                   // [2048,1536] f32
{
  const int lane = threadIdx.x & 31;
  const int wid  = threadIdx.x >> 5;
  const int wm   = wid >> 2, wn = wid & 3;
  const int g    = lane >> 4, l16 = lane & 15;

  const int m0 = blockIdx.y * 64 + wm * 32;
  const int n0 = blockIdx.x * 256 + wn * 64;

  const unsigned short* ap0 = Aemb + (size_t)(m0 + l16) * EK + 16 * g;
  const unsigned short* ap1 = Aemb + (size_t)(m0 + 16 + l16) * EK + 16 * g;
  const unsigned short* bp  = w1bf + (size_t)(n0 + l16) * EK + 16 * g;

  v8f acc[8] = {};
#pragma unroll
  for (int k = 0; k < EK; k += 32) {
    v16bf a0 = load_afrag(ap0 + k);
    v16bf a1 = load_afrag(ap1 + k);
    v16bf b0 = *reinterpret_cast<const v16bf*>(bp + k);
    v16bf b1 = *reinterpret_cast<const v16bf*>(bp + 16 * EK + k);
    v16bf b2 = *reinterpret_cast<const v16bf*>(bp + 32 * EK + k);
    v16bf b3 = *reinterpret_cast<const v16bf*>(bp + 48 * EK + k);
    acc[0] = wmma_bf16(a0, b0, acc[0]);
    acc[1] = wmma_bf16(a0, b1, acc[1]);
    acc[2] = wmma_bf16(a0, b2, acc[2]);
    acc[3] = wmma_bf16(a0, b3, acc[3]);
    acc[4] = wmma_bf16(a1, b0, acc[4]);
    acc[5] = wmma_bf16(a1, b1, acc[5]);
    acc[6] = wmma_bf16(a1, b2, acc[6]);
    acc[7] = wmma_bf16(a1, b3, acc[7]);
  }
#pragma unroll
  for (int j = 0; j < 4; ++j) {
    const int n = n0 + j * 16 + l16;
    const float bias = b_ih[n];
#pragma unroll
    for (int half = 0; half < 2; ++half) {
      const int mb = m0 + half * 16 + 8 * g;
#pragma unroll
      for (int r = 0; r < 8; ++r)
        gie[(size_t)(mb + r) * G3H + n] = acc[half * 4 + j][r] + bias;
    }
  }
}

// ---------------- recurrence: one workgroup per batch element ----------------

__global__ __launch_bounds__(1024) void k_recurrence(
    const float* __restrict__ enc,     // [B,S,H]
    const float* __restrict__ hinit,   // [B,H]
    const float* __restrict__ gie,     // [T*B, 3H] precomputed emb-part of gi
    const float* __restrict__ w_ihcT,  // [H, 3H]  (ctx columns of w_ih, transposed)
    const float* __restrict__ w_hhT,   // [H, 3H]
    const float* __restrict__ b_hh,    // [3H]
    const float* __restrict__ hm_wT,   // [H, H]
    const float* __restrict__ hm_b,    // [H]
    unsigned short* __restrict__ predA)// [T*B, 2H] bf16
{
  const int b   = blockIdx.x;
  const int tid = threadIdx.x;
  const int nth = blockDim.x;          // 1024

  __shared__ float h[Hsz];
  __shared__ float ctx[Hsz];
  __shared__ float gi[G3H];
  __shared__ float gh[G3H];
  __shared__ float sp[Ssz * 16];
  __shared__ float sc[Ssz];
  __shared__ float pw[Ssz];

  // ---- h0 = hm_w @ h_init[b] + hm_b (gi[] as temp for the input row) ----
  for (int k = tid; k < Hsz; k += nth) gi[k] = hinit[(size_t)b * Hsz + k];
  for (int k = tid; k < Hsz; k += nth) ctx[k] = 0.0f;
  __syncthreads();
  for (int j = tid; j < Hsz; j += nth) {
    float acc = hm_b[j];
    for (int k = 0; k < Hsz; ++k) acc += gi[k] * hm_wT[k * Hsz + j];
    h[j] = acc;
  }
  __syncthreads();

  const float* encb = enc + (size_t)b * Ssz * Hsz;

  for (int t = 0; t < Tsz; ++t) {
    const float* gier = gie + (size_t)(t * Bsz + b) * G3H;

    // ---- gi = gie_row + ctx @ w_ih[:,E:].T ;  gh = h @ w_hh.T + b_hh ----
    for (int j = tid; j < G3H; j += nth) {
      float acc = gier[j];
      for (int k = 0; k < Hsz; ++k) acc += ctx[k] * w_ihcT[k * G3H + j];
      gi[j] = acc;
    }
    for (int j = tid; j < G3H; j += nth) {
      float acc = b_hh[j];
      for (int k = 0; k < Hsz; ++k) acc += h[k] * w_hhT[k * G3H + j];
      gh[j] = acc;
    }
    __syncthreads();

    // ---- GRU gates, h update ----
    if (tid < Hsz) {
      float r  = 1.0f / (1.0f + __expf(-(gi[tid] + gh[tid])));
      float z  = 1.0f / (1.0f + __expf(-(gi[Hsz + tid] + gh[Hsz + tid])));
      float nn = tanhf(gi[2 * Hsz + tid] + r * gh[2 * Hsz + tid]);
      h[tid] = (1.0f - z) * nn + z * h[tid];
    }
    __syncthreads();

    // ---- attention scores: 16 threads per s ----
    {
      int s = tid >> 4, l = tid & 15;
      const float* er = encb + (size_t)s * Hsz;
      float acc = 0.0f;
      for (int k = l; k < Hsz; k += 16) acc += er[k] * h[k];
      sp[s * 16 + l] = acc;
    }
    __syncthreads();
    if (tid < Ssz) {
      float a = 0.0f;
      for (int i = 0; i < 16; ++i) a += sp[tid * 16 + i];
      sc[tid] = a;
    }
    __syncthreads();
    if (tid < Ssz) {
      float m = -1e30f;
      for (int s = 0; s < Ssz; ++s) m = fmaxf(m, sc[s]);
      float sum = 0.0f;
      for (int s = 0; s < Ssz; ++s) sum += __expf(sc[s] - m);
      pw[tid] = __expf(sc[tid] - m) / sum;
    }
    __syncthreads();

    // ---- ctx = sum_s p[s] * enc[b,s,:] ; emit pred row (bf16) ----
    if (tid < Hsz) {
      float acc = 0.0f;
      const float* eb = encb + tid;
      for (int s = 0; s < Ssz; ++s) acc += pw[s] * eb[(size_t)s * Hsz];
      ctx[tid] = acc;
      unsigned short* pr = predA + (size_t)(t * Bsz + b) * PK;
      pr[tid]       = f2bf(acc);
      pr[Hsz + tid] = f2bf(h[tid]);
    }
    __syncthreads();
  }
}

// ---------------- classifier: [2048,1024] x [1024,32000] bf16 WMMA ----------------
// block 256 thr (8 waves: 2M x 4N); wave tile 32M x 64N; block 64M x 256N
// grid = (32000/256, 2048/64) = (125, 32); K = 1024

__global__ __launch_bounds__(256) void k_classifier(
    const unsigned short* __restrict__ predA,  // [2048,1024] bf16
    const unsigned short* __restrict__ clsw,   // [32000,1024] bf16
    const float* __restrict__ clsb,            // [32000]
    float* __restrict__ out)                   // [B,T,V] f32
{
  const int lane = threadIdx.x & 31;
  const int wid  = threadIdx.x >> 5;
  const int wm   = wid >> 2, wn = wid & 3;
  const int g    = lane >> 4, l16 = lane & 15;

  const int m0 = blockIdx.y * 64 + wm * 32;
  const int n0 = blockIdx.x * 256 + wn * 64;

  const unsigned short* ap0 = predA + (size_t)(m0 + l16) * PK + 16 * g;
  const unsigned short* ap1 = predA + (size_t)(m0 + 16 + l16) * PK + 16 * g;
  const unsigned short* bp  = clsw + (size_t)(n0 + l16) * PK + 16 * g;

  v8f acc[8] = {};
#pragma unroll 2
  for (int k = 0; k < PK; k += 32) {
    v16bf a0 = load_afrag(ap0 + k);
    v16bf a1 = load_afrag(ap1 + k);
    v16bf b0 = *reinterpret_cast<const v16bf*>(bp + k);
    v16bf b1 = *reinterpret_cast<const v16bf*>(bp + 16 * PK + k);
    v16bf b2 = *reinterpret_cast<const v16bf*>(bp + 32 * PK + k);
    v16bf b3 = *reinterpret_cast<const v16bf*>(bp + 48 * PK + k);
    acc[0] = wmma_bf16(a0, b0, acc[0]);
    acc[1] = wmma_bf16(a0, b1, acc[1]);
    acc[2] = wmma_bf16(a0, b2, acc[2]);
    acc[3] = wmma_bf16(a0, b3, acc[3]);
    acc[4] = wmma_bf16(a1, b0, acc[4]);
    acc[5] = wmma_bf16(a1, b1, acc[5]);
    acc[6] = wmma_bf16(a1, b2, acc[6]);
    acc[7] = wmma_bf16(a1, b3, acc[7]);
  }

  // C/D layout: VGPR r -> (M = m0 + half*16 + 8g + r, N = n0 + j*16 + l16)
#pragma unroll
  for (int j = 0; j < 4; ++j) {
    const int n = n0 + j * 16 + l16;
    const float bias = clsb[n];
#pragma unroll
    for (int half = 0; half < 2; ++half) {
      const int mb = m0 + half * 16 + 8 * g;
#pragma unroll
      for (int r = 0; r < 8; ++r) {
        const int m  = mb + r;           // pred row = t*B + b
        const int tt = m >> 5;
        const int bb = m & 31;
        out[((size_t)bb * Tsz + tt) * Vsz + n] = acc[half * 4 + j][r] + bias;
      }
    }
  }
}

// ---------------- host ----------------

extern "C" void kernel_launch(void* const* d_in, const int* in_sizes, int n_in,
                              void* d_out, int out_size, void* d_ws, size_t ws_size,
                              hipStream_t stream) {
  const float* enc   = (const float*)d_in[0];
  const float* hinit = (const float*)d_in[1];
  const int*   tgt   = (const int*)d_in[2];
  const float* emb   = (const float*)d_in[3];
  const float* w_ih  = (const float*)d_in[4];
  const float* w_hh  = (const float*)d_in[5];
  const float* b_ih  = (const float*)d_in[6];
  const float* b_hh  = (const float*)d_in[7];
  const float* hm_w  = (const float*)d_in[8];
  const float* hm_b  = (const float*)d_in[9];
  const float* clsw  = (const float*)d_in[10];
  const float* clsb  = (const float*)d_in[11];
  float* out = (float*)d_out;

  char* ws = (char*)d_ws;
  size_t off = 0;
  unsigned short* clsw_bf = (unsigned short*)(ws + off); off += (size_t)Vsz * PK * 2;  // 65,536,000
  unsigned short* predA   = (unsigned short*)(ws + off); off += (size_t)PM * PK * 2;   //  4,194,304
  float* gie              = (float*)(ws + off);          off += (size_t)PM * G3H * 4;  // 12,582,912
  unsigned short* Aemb    = (unsigned short*)(ws + off); off += (size_t)PM * EK * 2;   //  1,048,576
  unsigned short* w1bf    = (unsigned short*)(ws + off); off += (size_t)G3H * EK * 2;  //    786,432
  float* w_ihcT           = (float*)(ws + off);          off += (size_t)Hsz * G3H * 4; //  3,145,728
  float* w_hhT            = (float*)(ws + off);          off += (size_t)Hsz * G3H * 4; //  3,145,728
  float* hm_wT            = (float*)(ws + off);          off += (size_t)Hsz * Hsz * 4; //  1,048,576

  // prep
  k_f32_to_bf16<<<2048, 256, 0, stream>>>(clsw, clsw_bf, (size_t)Vsz * PK);
  k_gather_emb<<<512, 256, 0, stream>>>(emb, tgt, Aemb);
  k_conv_sub<<<512, 256, 0, stream>>>(w_ih, w1bf, G3H, Esz + Hsz, 0, EK);
  k_transpose_sub<<<768, 256, 0, stream>>>(w_ih, w_ihcT, G3H, Esz + Hsz, Esz, Hsz);
  k_transpose_sub<<<768, 256, 0, stream>>>(w_hh, w_hhT, G3H, Hsz, 0, Hsz);
  k_transpose_sub<<<256, 256, 0, stream>>>(hm_w, hm_wT, Hsz, Hsz, 0, Hsz);

  // parallel WMMA: embedding part of the input gates for all (t,b)
  k_gie_gemm<<<dim3(6, 32), 256, 0, stream>>>(Aemb, w1bf, b_ih, gie);

  // sequential GRU+attention recurrence: batch-parallel, one WG per batch row
  k_recurrence<<<Bsz, 1024, 0, stream>>>(enc, hinit, gie, w_ihcT, w_hhT,
                                         b_hh, hm_wT, hm_b, predA);

  // fused classifier over all (t,b): [2048,1024] x [1024,32000] + bias
  k_classifier<<<dim3(125, 32), 256, 0, stream>>>(predA, clsw_bf, clsb, out);
}
